// Decoder_46669114638402
// MI455X (gfx1250) — matrix operand order repaired and checked by reading, not compiled, not run on hardware
//
#include <hip/hip_runtime.h>

// ---------- types ----------
typedef __attribute__((ext_vector_type(16))) __bf16 v16bf;
typedef __attribute__((ext_vector_type(8)))  __bf16 v8bf;
typedef __attribute__((ext_vector_type(8)))  float  v8f;

#define B_   32
#define S_   512
#define T_   256
#define KV_  128
#define EMB_ 256
#define H1_  512
#define VOC_ 32000
#define MTOT (B_ * T_)   // 8192

__device__ __forceinline__ float sigmoidf_(float x) {
  return 1.0f / (1.0f + __expf(-x));
}

__device__ __forceinline__ v8f zero_v8f() {
  v8f z = {0.f, 0.f, 0.f, 0.f, 0.f, 0.f, 0.f, 0.f};
  return z;
}

// Core: one wave computes a 16(M) x 64(N) f32 tile of  A(16xK) * B(Kx64).
// A: row-major, lda elements per row (bf16).
// Bc: "column n of B" == row n of Bc (row-major, ldb elements == K-contiguous).
// Software-pipelined one K-step ahead: loads for k+32 are issued before the
// wmmas for k, so each v_wmma waits on loads a full step old (staggered
// s_wait_loadcnt instead of load->wait0->wmma serialization).
template <int KDIM>
__device__ __forceinline__ void wmma_tile_16x64(const __bf16* __restrict__ A, int lda,
                                                const __bf16* __restrict__ Bc, int ldb,
                                                v8f acc[4]) {
  const int lane = threadIdx.x & 31;
  const int lm = lane & 15;       // row (A) / column (B) within 16
  const int lh = lane >> 4;       // half-wave selector
  const __bf16* a0 = A + (size_t)lm * lda + lh * 16;  // 16 contiguous K per lane
  const __bf16* b0 = Bc + (size_t)lm * ldb + lh * 16; // 16 contiguous K per lane

  union AV { v16bf v; v8bf h[2]; };
  AV a_cur, a_nxt;
  v16bf b_cur[4], b_nxt[4];

  a_cur.h[0] = *(const v8bf*)(a0);
  a_cur.h[1] = *(const v8bf*)(a0 + 8);
#pragma unroll
  for (int c = 0; c < 4; ++c)
    b_cur[c] = *(const v16bf*)(b0 + (size_t)(c * 16) * ldb);

#pragma unroll
  for (int k0 = 0; k0 < KDIM; k0 += 32) {
    if (k0 + 32 < KDIM) {  // compile-time under full unroll
      a_nxt.h[0] = *(const v8bf*)(a0 + k0 + 32);
      a_nxt.h[1] = *(const v8bf*)(a0 + k0 + 40);
#pragma unroll
      for (int c = 0; c < 4; ++c)
        b_nxt[c] = *(const v16bf*)(b0 + (size_t)(c * 16) * ldb + k0 + 32);
    }
#pragma unroll
    for (int c = 0; c < 4; ++c)
      acc[c] = __builtin_amdgcn_wmma_f32_16x16x32_bf16(
          false, a_cur.v, false, b_cur[c], (short)0, acc[c], false, false);
    a_cur = a_nxt;
#pragma unroll
    for (int c = 0; c < 4; ++c) b_cur[c] = b_nxt[c];
  }
}

// ---------- small prep kernels ----------
__global__ void cvt_bf16_kernel(const float* __restrict__ src, __bf16* __restrict__ dst, int n) {
  int i = blockIdx.x * blockDim.x + threadIdx.x;
  if (i < n) dst[i] = (__bf16)src[i];
}

// valT[b, v, s] = value[b, s, v]  (bf16), so context-GEMM B columns are K(s)-contiguous.
__global__ void transpose_value_kernel(const float* __restrict__ value, __bf16* __restrict__ valT) {
  int i = blockIdx.x * blockDim.x + threadIdx.x;      // B_*KV_*S_ threads
  int s = i & (S_ - 1);
  int v = (i >> 9) & (KV_ - 1);
  int b = i >> 16;
  valT[i] = (__bf16)value[((size_t)b * S_ + s) * KV_ + v];
}

// One wave per (b,t): gather embedding row (already bf16) for shifted targets.
__global__ void embed_kernel(const __bf16* __restrict__ Ebf, const int* __restrict__ y,
                             __bf16* __restrict__ X) {
  int wid = (blockIdx.x * blockDim.x + threadIdx.x) >> 5;  // row m = b*T + t
  int lane = threadIdx.x & 31;
  int t = wid & (T_ - 1);
  int b = wid >> 8;
  int idx = (t == 0) ? 0 : y[b * T_ + t - 1];
  const v8bf* src = (const v8bf*)(Ebf + (size_t)idx * EMB_);
  v8bf* dst = (v8bf*)(X + (size_t)wid * EMB_);
  dst[lane] = src[lane];                                   // 32 lanes x 16B = 256 bf16
}

// ---------- fused LSTM-cell GEMM (zero state: h = sig(o)*tanh(sig(i)*tanh(g))) ----------
// gates = A @ W^T + (b_ih+b_hh); W is (4H x K) row-major; f-gate dead.
template <int KDIM>
__global__ void lstm_gemm_kernel(const __bf16* __restrict__ A, const __bf16* __restrict__ W,
                                 const float* __restrict__ b_ih, const float* __restrict__ b_hh,
                                 __bf16* __restrict__ Hout, int Hdim, int ldh, int Mtiles) {
  int wid = (blockIdx.x * blockDim.x + threadIdx.x) >> 5;
  int mt = wid % Mtiles, nt = wid / Mtiles;
  int m0 = mt * 16, n0 = nt * 64;
  const __bf16* Am = A + (size_t)m0 * KDIM;
  v8f ai[4], ag[4], ao[4];
#pragma unroll
  for (int c = 0; c < 4; ++c) { ai[c] = zero_v8f(); ag[c] = zero_v8f(); ao[c] = zero_v8f(); }
  wmma_tile_16x64<KDIM>(Am, KDIM, W + (size_t)(0 * Hdim + n0) * KDIM, KDIM, ai); // i gate
  wmma_tile_16x64<KDIM>(Am, KDIM, W + (size_t)(2 * Hdim + n0) * KDIM, KDIM, ag); // g gate
  wmma_tile_16x64<KDIM>(Am, KDIM, W + (size_t)(3 * Hdim + n0) * KDIM, KDIM, ao); // o gate

  const int lane = threadIdx.x & 31, lm = lane & 15, lh = lane >> 4;
#pragma unroll
  for (int c = 0; c < 4; ++c) {
    int n = n0 + c * 16 + lm;
    float bi = b_ih[n] + b_hh[n];
    float bg = b_ih[2 * Hdim + n] + b_hh[2 * Hdim + n];
    float bo = b_ih[3 * Hdim + n] + b_hh[3 * Hdim + n];
#pragma unroll
    for (int r = 0; r < 8; ++r) {
      int m = m0 + r + 8 * lh;
      float cc = sigmoidf_(ai[c][r] + bi) * tanhf(ag[c][r] + bg);
      float h  = sigmoidf_(ao[c][r] + bo) * tanhf(cc);
      Hout[(size_t)m * ldh + n] = (__bf16)h;
    }
  }
}

// ---------- attention energy: energy[m, s] = h2[m,:] . key[b,s,:], masked ----------
__global__ void energy_kernel(const __bf16* __restrict__ ctx, const __bf16* __restrict__ keybf,
                              const int* __restrict__ enc_len, float* __restrict__ energy,
                              int Mtiles) {
  int wid = (blockIdx.x * blockDim.x + threadIdx.x) >> 5;
  int mt = wid % Mtiles, nt = wid / Mtiles;
  int m0 = mt * 16, n0 = nt * 64;
  int b = m0 >> 8;
  v8f acc[4];
#pragma unroll
  for (int c = 0; c < 4; ++c) acc[c] = zero_v8f();
  // A = out_ctx rows (first 128 cols = h2), ld 256; B col s = key[b,s,:] (128 contiguous)
  wmma_tile_16x64<KV_>(ctx + (size_t)m0 * EMB_, EMB_,
                       keybf + ((size_t)b * S_ + n0) * KV_, KV_, acc);
  int len = enc_len[b];
  const int lane = threadIdx.x & 31, lm = lane & 15, lh = lane >> 4;
#pragma unroll
  for (int c = 0; c < 4; ++c) {
    int s = n0 + c * 16 + lm;
#pragma unroll
    for (int r = 0; r < 8; ++r) {
      int m = m0 + r + 8 * lh;
      energy[(size_t)m * S_ + s] = (s >= len) ? -1e9f : acc[c][r];
    }
  }
}

// ---------- softmax over S per row (one wave per row), writes bf16 attn ----------
__global__ void softmax_kernel(const float* __restrict__ energy, __bf16* __restrict__ attn) {
  int wid = (blockIdx.x * blockDim.x + threadIdx.x) >> 5;
  int lane = threadIdx.x & 31;
  const float* row = energy + (size_t)wid * S_;
  float vals[S_ / 32];
  float mx = -3.4e38f;
#pragma unroll
  for (int j = 0; j < S_ / 32; ++j) { vals[j] = row[lane + 32 * j]; mx = fmaxf(mx, vals[j]); }
#pragma unroll
  for (int off = 16; off > 0; off >>= 1) mx = fmaxf(mx, __shfl_xor(mx, off, 32));
  float sum = 0.f;
#pragma unroll
  for (int j = 0; j < S_ / 32; ++j) { vals[j] = __expf(vals[j] - mx); sum += vals[j]; }
#pragma unroll
  for (int off = 16; off > 0; off >>= 1) sum += __shfl_xor(sum, off, 32);
  float inv = 1.f / sum;
  __bf16* arow = attn + (size_t)wid * S_;
#pragma unroll
  for (int j = 0; j < S_ / 32; ++j) arow[lane + 32 * j] = (__bf16)(vals[j] * inv);
}

// ---------- context = attn @ value, written into out_ctx[:, 128:256] ----------
__global__ void context_kernel(const __bf16* __restrict__ attn, const __bf16* __restrict__ valT,
                               __bf16* __restrict__ ctx, int Mtiles) {
  int wid = (blockIdx.x * blockDim.x + threadIdx.x) >> 5;
  int mt = wid % Mtiles, nt = wid / Mtiles;
  int m0 = mt * 16, n0 = nt * 64;
  int b = m0 >> 8;
  v8f acc[4];
#pragma unroll
  for (int c = 0; c < 4; ++c) acc[c] = zero_v8f();
  wmma_tile_16x64<S_>(attn + (size_t)m0 * S_, S_,
                      valT + ((size_t)b * KV_ + n0) * S_, S_, acc);
  const int lane = threadIdx.x & 31, lm = lane & 15, lh = lane >> 4;
#pragma unroll
  for (int c = 0; c < 4; ++c) {
    int n = n0 + c * 16 + lm;
#pragma unroll
    for (int r = 0; r < 8; ++r) {
      int m = m0 + r + 8 * lh;
      ctx[(size_t)m * EMB_ + KV_ + n] = (__bf16)acc[c][r];
    }
  }
}

// ---------- logits = out_ctx @ E^T + b_out, stored transposed (B, VOCAB, T) ----------
__global__ void logits_kernel(const __bf16* __restrict__ ctx, const __bf16* __restrict__ Ebf,
                              const float* __restrict__ b_out, float* __restrict__ out,
                              int Mtiles) {
  int wid = (blockIdx.x * blockDim.x + threadIdx.x) >> 5;
  int mt = wid % Mtiles, nt = wid / Mtiles;   // mt fastest: waves in flight share E panel
  int m0 = mt * 16, n0 = nt * 64;
  v8f acc[4];
#pragma unroll
  for (int c = 0; c < 4; ++c) acc[c] = zero_v8f();
  wmma_tile_16x64<EMB_>(ctx + (size_t)m0 * EMB_, EMB_,
                        Ebf + (size_t)n0 * EMB_, EMB_, acc);
  const int lane = threadIdx.x & 31, lm = lane & 15, lh = lane >> 4;
  int b = m0 >> 8;
  int tbase = (m0 & (T_ - 1)) + 8 * lh;       // 8 consecutive t per accumulator set
#pragma unroll
  for (int c = 0; c < 4; ++c) {
    int n = n0 + c * 16 + lm;
    float bo = b_out[n];
    float* po = out + ((size_t)b * VOC_ + n) * T_ + tbase;
    float4 lo = make_float4(acc[c][0] + bo, acc[c][1] + bo, acc[c][2] + bo, acc[c][3] + bo);
    float4 hi = make_float4(acc[c][4] + bo, acc[c][5] + bo, acc[c][6] + bo, acc[c][7] + bo);
    *(float4*)(po)     = lo;
    *(float4*)(po + 4) = hi;
  }
}

// ---------- host ----------
extern "C" void kernel_launch(void* const* d_in, const int* in_sizes, int n_in,
                              void* d_out, int out_size, void* d_ws, size_t ws_size,
                              hipStream_t stream) {
  const float* key   = (const float*)d_in[0];
  const float* value = (const float*)d_in[1];
  const int*   enc   = (const int*)d_in[2];
  const int*   y     = (const int*)d_in[3];
  const float* E     = (const float*)d_in[4];
  const float* W1    = (const float*)d_in[5];
  const float* bi1   = (const float*)d_in[6];
  const float* bh1   = (const float*)d_in[7];
  const float* W2    = (const float*)d_in[8];
  const float* bi2   = (const float*)d_in[9];
  const float* bh2   = (const float*)d_in[10];
  const float* bout  = (const float*)d_in[11];
  float* out = (float*)d_out;

  char* ws = (char*)d_ws;
  size_t off = 0;
  auto alloc = [&](size_t bytes) { size_t o = off; off += (bytes + 255) & ~(size_t)255; return o; };
  __bf16* Ebf   = (__bf16*)(ws + alloc((size_t)VOC_ * EMB_ * 2));   // 16 MB
  __bf16* W1bf  = (__bf16*)(ws + alloc((size_t)4 * H1_ * EMB_ * 2));
  __bf16* W2bf  = (__bf16*)(ws + alloc((size_t)4 * KV_ * H1_ * 2));
  __bf16* keybf = (__bf16*)(ws + alloc((size_t)B_ * S_ * KV_ * 2));
  __bf16* valT  = (__bf16*)(ws + alloc((size_t)B_ * KV_ * S_ * 2));
  __bf16* X     = (__bf16*)(ws + alloc((size_t)MTOT * EMB_ * 2));
  __bf16* h1bf  = (__bf16*)(ws + alloc((size_t)MTOT * H1_ * 2));
  __bf16* ctx   = (__bf16*)(ws + alloc((size_t)MTOT * EMB_ * 2));   // [h2 | context]
  float*  energy= (float*)(ws + alloc((size_t)MTOT * S_ * 4));      // 16 MB
  __bf16* attn  = (__bf16*)(ws + alloc((size_t)MTOT * S_ * 2));
  (void)ws_size;

  const int TB = 256;  // 8 waves / block everywhere

  // 1) precision conversion (one pass; weights stay L2-resident afterwards)
  { int n = VOC_ * EMB_;        cvt_bf16_kernel<<<(n + TB - 1) / TB, TB, 0, stream>>>(E,  Ebf,  n); }
  { int n = 4 * H1_ * EMB_;     cvt_bf16_kernel<<<(n + TB - 1) / TB, TB, 0, stream>>>(W1, W1bf, n); }
  { int n = 4 * KV_ * H1_;      cvt_bf16_kernel<<<(n + TB - 1) / TB, TB, 0, stream>>>(W2, W2bf, n); }
  { int n = B_ * S_ * KV_;      cvt_bf16_kernel<<<(n + TB - 1) / TB, TB, 0, stream>>>(key, keybf, n); }
  { int n = B_ * KV_ * S_;      transpose_value_kernel<<<n / TB, TB, 0, stream>>>(value, valT); }

  // 2) embedding gather for shifted targets (wave per row)
  embed_kernel<<<MTOT / 8, TB, 0, stream>>>(Ebf, y, X);

  // 3) LSTM1: 8192x2048 gate GEMM fused with activations -> h1 (bf16)
  { int waves = (MTOT / 16) * (H1_ / 64);
    lstm_gemm_kernel<EMB_><<<waves / 8, TB, 0, stream>>>(X, W1bf, bi1, bh1, h1bf, H1_, H1_, MTOT / 16); }

  // 4) LSTM2 -> h2 written into out_ctx[:, 0:128] (bf16)
  { int waves = (MTOT / 16) * (KV_ / 64);
    lstm_gemm_kernel<H1_><<<waves / 8, TB, 0, stream>>>(h1bf, W2bf, bi2, bh2, ctx, KV_, EMB_, MTOT / 16); }

  // 5) attention energy (masked), softmax, context -> out_ctx[:, 128:256]
  { int waves = (MTOT / 16) * (S_ / 64);
    energy_kernel<<<waves / 8, TB, 0, stream>>>(ctx, keybf, enc, energy, MTOT / 16); }
  softmax_kernel<<<MTOT / 8, TB, 0, stream>>>(energy, attn);
  { int waves = (MTOT / 16) * (KV_ / 64);
    context_kernel<<<waves / 8, TB, 0, stream>>>(attn, valT, ctx, MTOT / 16); }

  // 6) tied-embedding logits GEMM (134 GF, HBM-write-bound), transposed store
  { int waves = (MTOT / 16) * (VOC_ / 64);
    logits_kernel<<<waves / 8, TB, 0, stream>>>(ctx, Ebf, bout, out, MTOT / 16); }
  (void)in_sizes; (void)n_in; (void)out_size;
}
